// CausalSelfAttention_38620345926298
// MI455X (gfx1250) — compile-verified, matrix operand order checked by reading, not run on hardware
//
#include <hip/hip_runtime.h>
#include <hip/hip_bf16.h>

// ---------------- problem constants ----------------
#define T_SEQ 827
#define TP    832            // T padded to 52*16
#define CDIM  1024
#define NHEAD 16
#define HD    64
#define NEG_BIG (-1e30f)

typedef __attribute__((ext_vector_type(16))) __bf16 bf16x16;
typedef __attribute__((ext_vector_type(8)))  float  f32x8;
typedef int v4i __attribute__((vector_size(16)));   // matches builtin param type

// Async global->LDS path (CDNA5); guarded so either toolchain compiles.
#if defined(__has_builtin)
#if __has_builtin(__builtin_amdgcn_global_load_async_to_lds_b128)
#define ASYNC_LDS 1
#endif
#endif

__device__ __forceinline__ void copy16_to_lds(void* l, const void* g) {
#ifdef ASYNC_LDS
  __builtin_amdgcn_global_load_async_to_lds_b128(
      (__attribute__((address_space(1))) v4i*)g,
      (__attribute__((address_space(3))) v4i*)l, 0, 0);
#else
  *(uint4*)l = *(const uint4*)g;
#endif
}

__device__ __forceinline__ void wait_async_stage() {
#ifdef ASYNC_LDS
#if __has_builtin(__builtin_amdgcn_s_wait_asynccnt)
  __builtin_amdgcn_s_wait_asynccnt(0);
#else
  asm volatile("s_wait_asynccnt 0" ::: "memory");
#endif
#endif
}

// 16-bit WMMA fragment (ISA 7.12.2, wave32): per lane the 16 elements are
// two contiguous 16B chunks of the source row:  [8h .. 8h+7] and [16+8h .. 23+8h].
union FragU { bf16x16 v; uint4 q[2]; };
__device__ __forceinline__ bf16x16 frag_row(const __bf16* row, int half) {
  FragU f;
  f.q[0] = *(const uint4*)(row + 8 * half);
  f.q[1] = *(const uint4*)(row + 16 + 8 * half);
  return f.v;
}

__device__ __forceinline__ void store4_bf16(__bf16* dst, float a, float b,
                                            float c, float d) {
  union { __bf16 h[4]; unsigned long long u; } pk;
  pk.h[0] = (__bf16)a; pk.h[1] = (__bf16)b; pk.h[2] = (__bf16)c; pk.h[3] = (__bf16)d;
  *(unsigned long long*)dst = pk.u;   // ds_store_b64
}

// ---------------------------------------------------------------------------
// Projection GEMM: 64(M)x64(N) tile per 128-thr block, K stepped by 32.
// A rows (x) row-major in LDS; B (W) stored TRANSPOSED in LDS so B-fragments
// are contiguous b128 loads. Boundary rows clamped (dups of row T-1; padded
// outputs are never consumed).
// TRANSPOSED=false: dst[b,nh,t,hd] bf16 (scatter).  TRANSPOSED=true (for V):
// dst[b,nh,hd,t] bf16, epilogue packs 8 consecutive t into one b128 store.
// ---------------------------------------------------------------------------
template <bool TRANSPOSED>
__global__ __launch_bounds__(128)
void proj_qkv_kernel(const float* __restrict__ x, const float* __restrict__ W,
                     const float* __restrict__ bias, __bf16* __restrict__ dst,
                     float scale)
{
  __shared__ __align__(16) __bf16 As[64][32];
  __shared__ __align__(16) __bf16 BsT[64][32];
  const int mt = blockIdx.x, nt = blockIdx.y, b = blockIdx.z;
  const int tid = threadIdx.x;
  const int w = tid >> 5, lane = tid & 31, r = lane & 15, half = lane >> 4;
  const int t0 = mt * 64, n0 = nt * 64;
  const float* xb = x + (size_t)b * T_SEQ * CDIM;

  f32x8 acc[4];
  #pragma unroll
  for (int j = 0; j < 4; ++j)
    #pragma unroll
    for (int i = 0; i < 8; ++i) acc[j][i] = 0.f;

  for (int kb = 0; kb < CDIM / 32; ++kb) {
    const int k0 = kb * 32;
    #pragma unroll
    for (int e = 0; e < 4; ++e) {
      int idx4 = tid + 128 * e;                 // 512 float4 per tile
      // x tile: 64 rows x 32 cols  (8 float4 per row)
      int mm = idx4 >> 3, kk = (idx4 & 7) * 4;
      int tc = min(t0 + mm, T_SEQ - 1);         // clamp, no branch
      float4 xv = *(const float4*)&xb[(size_t)tc * CDIM + k0 + kk];
      store4_bf16(&As[mm][kk], xv.x, xv.y, xv.z, xv.w);
      // W tile: 32 rows x 64 cols (16 float4 per row) -> transposed into BsT
      int kk2 = idx4 >> 4, nn = (idx4 & 15) * 4;
      float4 wv = *(const float4*)&W[(size_t)(k0 + kk2) * CDIM + n0 + nn];
      BsT[nn + 0][kk2] = (__bf16)wv.x;
      BsT[nn + 1][kk2] = (__bf16)wv.y;
      BsT[nn + 2][kk2] = (__bf16)wv.z;
      BsT[nn + 3][kk2] = (__bf16)wv.w;
    }
    __syncthreads();

    bf16x16 aF = frag_row(&As[w * 16 + r][0], half);
    #pragma unroll
    for (int j = 0; j < 4; ++j) {
      bf16x16 bF = frag_row(&BsT[j * 16 + r][0], half);
      acc[j] = __builtin_amdgcn_wmma_f32_16x16x32_bf16(false, aF, false, bF,
                                                       (short)0, acc[j], false, false);
    }
    __syncthreads();
  }

  #pragma unroll
  for (int j = 0; j < 4; ++j) {
    int c = n0 + j * 16 + r;
    int nh = c >> 6, hd = c & 63;
    float bv = bias[c];
    if (TRANSPOSED) {
      union { __bf16 h[8]; uint4 u; } pk;
      #pragma unroll
      for (int i = 0; i < 8; ++i) pk.h[i] = (__bf16)((acc[j][i] + bv) * scale);
      __bf16* dp = dst + (((size_t)b * NHEAD + nh) * HD + hd) * TP
                       + t0 + w * 16 + 8 * half;        // 8 consecutive t
      *(uint4*)dp = pk.u;
    } else {
      #pragma unroll
      for (int i = 0; i < 8; ++i) {
        int t = t0 + w * 16 + i + 8 * half;
        dst[(((size_t)b * NHEAD + nh) * TP + t) * HD + hd] =
            (__bf16)((acc[j][i] + bv) * scale);
      }
    }
  }
}

// ---------------------------------------------------------------------------
// Output projection: out[b,t,c] = (ybf @ Wp) + bp   (f32, rows t<T)
// Epilogue split: uniform (scalar-branch) fast path for full tiles; the
// per-element guard only exists in the single boundary M-tile.
// ---------------------------------------------------------------------------
__global__ __launch_bounds__(128)
void proj_out_kernel(const __bf16* __restrict__ y, const float* __restrict__ W,
                     const float* __restrict__ bias, float* __restrict__ out)
{
  __shared__ __align__(16) __bf16 As[64][32];
  __shared__ __align__(16) __bf16 BsT[64][32];
  const int mt = blockIdx.x, nt = blockIdx.y, b = blockIdx.z;
  const int tid = threadIdx.x;
  const int w = tid >> 5, lane = tid & 31, r = lane & 15, half = lane >> 4;
  const int t0 = mt * 64, n0 = nt * 64;
  const __bf16* yb = y + (size_t)b * TP * CDIM;

  f32x8 acc[4];
  #pragma unroll
  for (int j = 0; j < 4; ++j)
    #pragma unroll
    for (int i = 0; i < 8; ++i) acc[j][i] = 0.f;

  for (int kb = 0; kb < CDIM / 32; ++kb) {
    const int k0 = kb * 32;
    #pragma unroll
    for (int e = 0; e < 2; ++e) {
      // y tile: 64x32 bf16 = 256 uint4 (4 per row)
      int idx = tid + 128 * e;
      int mm = idx >> 2, kk = (idx & 3) * 8;
      int tc = min(t0 + mm, T_SEQ - 1);
      *(uint4*)&As[mm][kk] = *(const uint4*)&yb[(size_t)tc * CDIM + k0 + kk];
    }
    #pragma unroll
    for (int e = 0; e < 4; ++e) {
      int idx4 = tid + 128 * e;
      int kk2 = idx4 >> 4, nn = (idx4 & 15) * 4;
      float4 wv = *(const float4*)&W[(size_t)(k0 + kk2) * CDIM + n0 + nn];
      BsT[nn + 0][kk2] = (__bf16)wv.x;
      BsT[nn + 1][kk2] = (__bf16)wv.y;
      BsT[nn + 2][kk2] = (__bf16)wv.z;
      BsT[nn + 3][kk2] = (__bf16)wv.w;
    }
    __syncthreads();

    bf16x16 aF = frag_row(&As[w * 16 + r][0], half);
    #pragma unroll
    for (int j = 0; j < 4; ++j) {
      bf16x16 bF = frag_row(&BsT[j * 16 + r][0], half);
      acc[j] = __builtin_amdgcn_wmma_f32_16x16x32_bf16(false, aF, false, bF,
                                                       (short)0, acc[j], false, false);
    }
    __syncthreads();
  }

  float* ob = out + (size_t)b * T_SEQ * CDIM;
  if (t0 + 64 <= T_SEQ) {                 // uniform fast path: no guards at all
    #pragma unroll
    for (int j = 0; j < 4; ++j) {
      int c = n0 + j * 16 + r;
      float bv = bias[c];
      #pragma unroll
      for (int i = 0; i < 8; ++i) {
        int t = t0 + w * 16 + i + 8 * half;
        ob[(size_t)t * CDIM + c] = acc[j][i] + bv;
      }
    }
  } else {                                // single boundary tile
    #pragma unroll
    for (int j = 0; j < 4; ++j) {
      int c = n0 + j * 16 + r;
      float bv = bias[c];
      #pragma unroll
      for (int i = 0; i < 8; ++i) {
        int t = t0 + w * 16 + i + 8 * half;
        if (t < T_SEQ) ob[(size_t)t * CDIM + c] = acc[j][i] + bv;
      }
    }
  }
}

// ---------------------------------------------------------------------------
// Blur pipeline (tiny fraction of runtime)
// ---------------------------------------------------------------------------
__global__ void blur_pass1(const float* __restrict__ f, const float* __restrict__ bmat,
                           float* __restrict__ blurred, unsigned int* __restrict__ mnmx)
{
  __shared__ float tile[16][16];
  const int blk = blockIdx.x;          // b*256 + i
  const int b = blk >> 8, i = blk & 255;
  const int v = threadIdx.x & 15, u = threadIdx.x >> 4;
  const int j = u * 16 + v;

  float val = f[((size_t)b * 256 + i) * 256 + j] *
              bmat[((size_t)b * 256 + j) * 256 + i];   // b^T
  tile[u][v] = val;
  __syncthreads();

  float k1[7]; float s = 0.f;
  #pragma unroll
  for (int d = 0; d < 7; ++d) { float z = (d - 3) / 1.5f; k1[d] = __expf(-0.5f * z * z); s += k1[d]; }
  const float inv = 1.f / s;
  #pragma unroll
  for (int d = 0; d < 7; ++d) k1[d] *= inv;

  float acc = 0.f;
  #pragma unroll
  for (int du = 0; du < 7; ++du) {
    int uu = u + du - 3; uu = uu < 0 ? -uu : (uu > 15 ? 30 - uu : uu);
    float rowacc = 0.f;
    #pragma unroll
    for (int dv = 0; dv < 7; ++dv) {
      int vv = v + dv - 3; vv = vv < 0 ? -vv : (vv > 15 ? 30 - vv : vv);
      rowacc += k1[dv] * tile[uu][vv];
    }
    acc += k1[du] * rowacc;
  }
  blurred[(size_t)blk * 256 + j] = acc;

  float mn = acc, mx = acc;
  #pragma unroll
  for (int d = 16; d >= 1; d >>= 1) {
    mn = fminf(mn, __shfl_xor(mn, d, 32));
    mx = fmaxf(mx, __shfl_xor(mx, d, 32));
  }
  if ((threadIdx.x & 31) == 0) {
    atomicMin(&mnmx[0], __float_as_uint(mn));   // all values >= 0: uint order ok
    atomicMax(&mnmx[1], __float_as_uint(mx));
  }
}

__global__ void blur_pass2(const float* __restrict__ f, const float* __restrict__ bmat,
                           float* __restrict__ blurmap, const unsigned int* __restrict__ mnmx)
{
  const int idx = blockIdx.x * blockDim.x + threadIdx.x;   // B*256*256
  const int b = idx >> 16;
  const int ij = idx & 65535;
  const int i = ij >> 8, j = ij & 255;
  const float mn = __uint_as_float(mnmx[0]);
  const float mx = __uint_as_float(mnmx[1]);
  float w = (blurmap[idx] - mn) / (mx - mn);
  w = fminf(fmaxf(w, 0.f), 1.f);
  blurmap[idx] = w * f[idx] * bmat[((size_t)b * 256 + j) * 256 + i];
  (void)i;
}

__global__ void init_minmax(unsigned int* mnmx) {
  if (threadIdx.x < 6) mnmx[threadIdx.x] = (threadIdx.x & 1) ? 0u : 0x7F800000u;
}

// ---------------------------------------------------------------------------
// Flash attention. 4 waves/WG (16 q-rows each), 32-wide s-blocks, DOUBLE-
// BUFFERED K/V staging (async-to-LDS when available). V is pre-transposed in
// the workspace so both K and V fragments are contiguous b128 LDS loads.
// Bias/blur application is fully branchless (clamped addresses + selects).
// ---------------------------------------------------------------------------
__global__ __launch_bounds__(128)
void attn_kernel(const __bf16* __restrict__ qb, const __bf16* __restrict__ kb,
                 const __bf16* __restrict__ vbT, const float* __restrict__ hbias,
                 const float* __restrict__ bl01, const float* __restrict__ bl02,
                 const float* __restrict__ bl12, __bf16* __restrict__ ybf)
{
  __shared__ __align__(16) __bf16 ks[2][32][HD];    // (s, d) row-major
  __shared__ __align__(16) __bf16 vsT[2][HD][32];   // (d, s) row-major
  __shared__ __align__(16) __bf16 ps[4][16][32];    // per-wave P bounce

  const int qt = blockIdx.x, hh = blockIdx.y, b = blockIdx.z;
  const int tid = threadIdx.x;
  const int w = tid >> 5, lane = tid & 31, r = lane & 15, half = lane >> 4;
  const int t0 = qt * 64;
  const int tw0 = t0 + w * 16;

  const __bf16* qp = qb  + ((size_t)(b * NHEAD + hh) * TP) * HD;
  const __bf16* kp = kb  + ((size_t)(b * NHEAD + hh) * TP) * HD;
  const __bf16* vTp = vbT + ((size_t)(b * NHEAD + hh) * HD) * TP;
  const float*  hp = hbias + ((size_t)(b * NHEAD + hh) * T_SEQ) * T_SEQ;

  // Q fragments: 2 x (global_load_b128 x2)
  const __bf16* qrow = qp + (size_t)(tw0 + r) * HD;
  bf16x16 qA0 = frag_row(qrow, half);
  bf16x16 qA1 = frag_row(qrow + 32, half);

  float m_i[8], l_i[8];
  f32x8 o[4];
  #pragma unroll
  for (int i = 0; i < 8; ++i) { m_i[i] = NEG_BIG; l_i[i] = 0.f; }
  #pragma unroll
  for (int j = 0; j < 4; ++j)
    #pragma unroll
    for (int i = 0; i < 8; ++i) o[j][i] = 0.f;

  const int maxT = min(t0 + 64, T_SEQ);
  const int nS = (maxT + 31) >> 5;

  // stage s-block sb into buffer `bf` (256+256 uint4, 4 per thread)
  auto stage = [&](int sb, int bf) {
    const int s0 = sb * 32;
    const __bf16* gk = kp + (size_t)s0 * HD;
    #pragma unroll
    for (int e = 0; e < 2; ++e) {
      int idx = tid + 128 * e;
      copy16_to_lds(&ks[bf][0][0] + idx * 8, gk + idx * 8);
      int row = idx >> 2, col = (idx & 3) * 8;
      copy16_to_lds(&vsT[bf][row][col], vTp + (size_t)row * TP + s0 + col);
    }
  };

  int buf = 0;
  stage(0, 0);
  for (int sb = 0; sb < nS; ++sb) {
    const int s0 = sb * 32;
    wait_async_stage();
    __syncthreads();                      // staged data visible; prev compute done
    if (sb + 1 < nS) stage(sb + 1, buf ^ 1);

    // ---- scores: two 16-col subtiles ----
    f32x8 attv[2];
    #pragma unroll
    for (int sj = 0; sj < 2; ++sj) {
      const __bf16* krow = &ks[buf][sj * 16 + r][0];
      bf16x16 kB0 = frag_row(krow, half);
      bf16x16 kB1 = frag_row(krow + 32, half);
      f32x8 acc;
      #pragma unroll
      for (int i = 0; i < 8; ++i) acc[i] = 0.f;
      acc = __builtin_amdgcn_wmma_f32_16x16x32_bf16(false, qA0, false, kB0,
                                                    (short)0, acc, false, false);
      acc = __builtin_amdgcn_wmma_f32_16x16x32_bf16(false, qA1, false, kB1,
                                                    (short)0, acc, false, false);

      const int s = s0 + sj * 16 + r;
      const int sc = min(s, T_SEQ - 1);
      #pragma unroll
      for (int i = 0; i < 8; ++i) {
        const int t = tw0 + i + 8 * half;
        const int tc = min(t, T_SEQ - 1);
        float a = acc[i] + hp[(size_t)tc * T_SEQ + sc];  // masked if s>t anyway
        // branchless epipolar blur multipliers (maps are L2-resident)
        int r01 = min(max(t - 285, 0), 255);
        int r2x = min(max(t - 571, 0), 255);
        int c12 = min(max(s - 286, 0), 255);
        float m01 = bl01[(((size_t)b << 8) + r01) * 256 + min(s, 255)];
        float m02 = bl02[(((size_t)b << 8) + r2x) * 256 + min(s, 255)];
        float m12 = bl12[(((size_t)b << 8) + r2x) * 256 + c12];
        bool in01 = (t >= 285) & (t < 541) & (s < 256);
        bool in02 = (t >= 571) & (t < T_SEQ) & (s < 256);
        bool in12 = (t >= 571) & (t < T_SEQ) & (s >= 286) & (s < 542);
        a *= in01 ? m01 : 1.f;
        a *= in02 ? m02 : 1.f;
        a *= in12 ? m12 : 1.f;
        a = (s > t) ? NEG_BIG : a;                        // causal
        attv[sj][i] = a;
      }
    }

    // ---- online softmax (row = 16 lanes of one half-wave) ----
    #pragma unroll
    for (int i = 0; i < 8; ++i) {
      float rm = fmaxf(attv[0][i], attv[1][i]);
      #pragma unroll
      for (int d = 8; d >= 1; d >>= 1) rm = fmaxf(rm, __shfl_xor(rm, d, 32));
      float mnew = fmaxf(m_i[i], rm);
      float corr = __expf(m_i[i] - mnew);
      float p0 = __expf(attv[0][i] - mnew);
      float p1 = __expf(attv[1][i] - mnew);
      float rs = p0 + p1;
      #pragma unroll
      for (int d = 8; d >= 1; d >>= 1) rs += __shfl_xor(rs, d, 32);
      l_i[i] = l_i[i] * corr + rs;
      m_i[i] = mnew;
      #pragma unroll
      for (int j = 0; j < 4; ++j) o[j][i] *= corr;
      ps[w][i + 8 * half][r]      = (__bf16)p0;
      ps[w][i + 8 * half][16 + r] = (__bf16)p1;
    }

    // ---- O += P(16x32) @ V(32x64) ----
    bf16x16 pA = frag_row(&ps[w][r][0], half);   // same-wave LDS RAW
    #pragma unroll
    for (int j = 0; j < 4; ++j) {
      bf16x16 vB = frag_row(&vsT[buf][j * 16 + r][0], half);
      o[j] = __builtin_amdgcn_wmma_f32_16x16x32_bf16(false, pA, false, vB,
                                                     (short)0, o[j], false, false);
    }
    buf ^= 1;
  }

  // normalize and store y (bf16, (B,TP,C)); uniform fast path off the boundary
  __bf16* yb = ybf + (size_t)b * TP * CDIM;
  if (t0 + 64 <= T_SEQ) {
    #pragma unroll
    for (int i = 0; i < 8; ++i) {
      int t = tw0 + i + 8 * half;
      float invl = 1.0f / l_i[i];
      #pragma unroll
      for (int j = 0; j < 4; ++j) {
        int c = hh * HD + j * 16 + r;
        yb[(size_t)t * CDIM + c] = (__bf16)(o[j][i] * invl);
      }
    }
  } else {
    #pragma unroll
    for (int i = 0; i < 8; ++i) {
      int t = tw0 + i + 8 * half;
      if (t < T_SEQ) {
        float invl = 1.0f / l_i[i];
        #pragma unroll
        for (int j = 0; j < 4; ++j) {
          int c = hh * HD + j * 16 + r;
          yb[(size_t)t * CDIM + c] = (__bf16)(o[j][i] * invl);
        }
      }
    }
  }
}

// ---------------------------------------------------------------------------
// launch
// ---------------------------------------------------------------------------
extern "C" void kernel_launch(void* const* d_in, const int* in_sizes, int n_in,
                              void* d_out, int out_size, void* d_ws, size_t ws_size,
                              hipStream_t stream)
{
  (void)in_sizes; (void)n_in; (void)out_size; (void)ws_size;
  const float* x   = (const float*)d_in[0];
  const float* h   = (const float*)d_in[1];
  const float* f01 = (const float*)d_in[2];
  const float* f02 = (const float*)d_in[3];
  const float* f12 = (const float*)d_in[4];
  const float* b01 = (const float*)d_in[5];
  const float* b02 = (const float*)d_in[6];
  const float* b12 = (const float*)d_in[7];
  const float* Wq  = (const float*)d_in[8];
  const float* bq  = (const float*)d_in[9];
  const float* Wk  = (const float*)d_in[10];
  const float* bk  = (const float*)d_in[11];
  const float* Wv  = (const float*)d_in[12];
  const float* bv  = (const float*)d_in[13];
  const float* Wp  = (const float*)d_in[14];
  const float* bp  = (const float*)d_in[15];
  float* out = (float*)d_out;

  char* ws = (char*)d_ws;
  __bf16* qb  = (__bf16*)(ws + 0);          // (B,NH,TP,HD) bf16
  __bf16* kb  = (__bf16*)(ws + 6815744);    // (B,NH,TP,HD) bf16
  __bf16* vbT = (__bf16*)(ws + 13631488);   // (B,NH,HD,TP) bf16  (transposed!)
  __bf16* ybf = (__bf16*)(ws + 20447232);   // (B,TP,C) bf16
  float* bl01 = (float*)(ws + 27262976);
  float* bl02 = (float*)(ws + 28311552);
  float* bl12 = (float*)(ws + 29360128);
  unsigned int* mnmx = (unsigned int*)(ws + 30408704);

  init_minmax<<<1, 32, 0, stream>>>(mnmx);
  blur_pass1<<<1024, 256, 0, stream>>>(f01, b01, bl01, mnmx + 0);
  blur_pass1<<<1024, 256, 0, stream>>>(f02, b02, bl02, mnmx + 2);
  blur_pass1<<<1024, 256, 0, stream>>>(f12, b12, bl12, mnmx + 4);
  blur_pass2<<<1024, 256, 0, stream>>>(f01, b01, bl01, mnmx + 0);
  blur_pass2<<<1024, 256, 0, stream>>>(f02, b02, bl02, mnmx + 2);
  blur_pass2<<<1024, 256, 0, stream>>>(f12, b12, bl12, mnmx + 4);

  dim3 gP(13, 16, 4);   // (TP/64, C/64, B)
  proj_qkv_kernel<false><<<gP, 128, 0, stream>>>(x, Wq, bq, qb, 0.125f);
  proj_qkv_kernel<false><<<gP, 128, 0, stream>>>(x, Wk, bk, kb, 1.0f);
  proj_qkv_kernel<true ><<<gP, 128, 0, stream>>>(x, Wv, bv, vbT, 1.0f);

  attn_kernel<<<dim3(13, 16, 4), 128, 0, stream>>>(qb, kb, vbT, h,
                                                   bl01, bl02, bl12, ybf);

  proj_out_kernel<<<dim3(13, 16, 4), 128, 0, stream>>>(ybf, Wp, bp, out);
}
